// Head_81028853007035
// MI455X (gfx1250) — matrix operand order repaired and checked by reading, not compiled, not run on hardware
//
#include <hip/hip_runtime.h>
#include <hip/hip_bf16.h>

// ---------------------------------------------------------------------------
// Causal single-head attention, flash-style, f16 WMMA with f32 accumulation.
// B=512, T=256, C=384, H=64.
// ---------------------------------------------------------------------------

typedef _Float16 half8  __attribute__((ext_vector_type(8)));
typedef _Float16 v16h   __attribute__((ext_vector_type(16)));
typedef float    v8f    __attribute__((ext_vector_type(8)));
typedef float    f4     __attribute__((ext_vector_type(4)));

#define WMMA_F16(a, b, c) \
    __builtin_amdgcn_wmma_f32_16x16x32_f16(false, (a), false, (b), (short)0, (c), false, false)

static constexpr int BATCH = 512;
static constexpr int TSEQ  = 256;
static constexpr int CEMB  = 384;
static constexpr int HSZ   = 64;

// ws layout (in _Float16 units)
static constexpr size_t WT_OFF  = 0;                       // 3 * 64*384
static constexpr size_t WT_ELEM = 3 * 64 * 384;            // 73728
static constexpr size_t QKV_ELEM = (size_t)BATCH * TSEQ * HSZ;  // 8388608
static constexpr size_t Q_OFF = WT_ELEM;
static constexpr size_t K_OFF = Q_OFF + QKV_ELEM;
static constexpr size_t V_OFF = K_OFF + QKV_ELEM;

// ---------------------------------------------------------------------------
// Kernel 1: convert + transpose weights to f16:  WT[mat][n*384 + c] = W[c][n]
// mat: 0 = Wq, 1 = Wk, 2 = Wv   (contiguous-in-c so WMMA B loads are 32B runs)
// ---------------------------------------------------------------------------
__global__ __launch_bounds__(256)
void wt_convert_kernel(const float* __restrict__ Wk, const float* __restrict__ Wq,
                       const float* __restrict__ Wv, _Float16* __restrict__ WT) {
    int id = blockIdx.x * 256 + threadIdx.x;          // 0 .. 73727
    int mat = id / (64 * 384);
    int rem = id - mat * (64 * 384);
    int n = rem / 384;
    int c = rem - n * 384;
    const float* W = (mat == 0) ? Wq : (mat == 1) ? Wk : Wv;
    WT[id] = (_Float16)W[c * 64 + n];
}

// ---------------------------------------------------------------------------
// Kernel 2: QKV projection.  One wave owns 16 rows of M = B*T.
// 12 output tiles (Q/K/V x 4 n-tiles of 16), K-loop of 12 chunks of 32.
// Q,K stored [row][h] f16;  V stored transposed Vt[b][h][s] f16.
// ---------------------------------------------------------------------------
__global__ __launch_bounds__(256)
void qkv_proj_kernel(const float* __restrict__ x, const _Float16* __restrict__ WT,
                     _Float16* __restrict__ Qh, _Float16* __restrict__ Kh,
                     _Float16* __restrict__ Vt) {
    const int lane = threadIdx.x & 31;
    const int wv   = threadIdx.x >> 5;
    const int l16  = lane & 15;
    const int hf   = lane >> 4;
    const int row_base = (blockIdx.x * 8 + wv) * 16;

    v8f acc[12] = {};

    const float* xrow = x + (size_t)(row_base + l16) * CEMB;

    #pragma unroll
    for (int kc = 0; kc < 12; ++kc) {
        // A (x tile, 16x32): lane holds row l16; elem i -> c = kc*32 + (i<8 ? hf*8+i : 16+hf*8+i-8)
        const int c0 = kc * 32 + hf * 8;
        f4 a0 = *(const f4*)(xrow + c0);
        f4 a1 = *(const f4*)(xrow + c0 + 4);
        f4 b0 = *(const f4*)(xrow + c0 + 16);
        f4 b1 = *(const f4*)(xrow + c0 + 20);
        v16h A = { (_Float16)a0[0], (_Float16)a0[1], (_Float16)a0[2], (_Float16)a0[3],
                   (_Float16)a1[0], (_Float16)a1[1], (_Float16)a1[2], (_Float16)a1[3],
                   (_Float16)b0[0], (_Float16)b0[1], (_Float16)b0[2], (_Float16)b0[3],
                   (_Float16)b1[0], (_Float16)b1[1], (_Float16)b1[2], (_Float16)b1[3] };

        #pragma unroll
        for (int j = 0; j < 12; ++j) {
            const int mat = j >> 2, nt = j & 3;
            // B (W tile, 32x16): lane holds n = nt*16+l16; elem i -> c = kc*32 + (i<8 ? hf*8+i : 16+hf*8+i-8)
            const _Float16* wp = WT + (size_t)mat * (64 * 384)
                               + (size_t)(nt * 16 + l16) * 384 + kc * 32 + hf * 8;
            half8 blo = *(const half8*)wp;
            half8 bhi = *(const half8*)(wp + 16);
            v16h B = __builtin_shufflevector(blo, bhi,
                        0,1,2,3,4,5,6,7,8,9,10,11,12,13,14,15);
            acc[j] = WMMA_F16(A, B, acc[j]);
        }
    }

    // stores. C layout: VGPR r -> row (r + 8*hf), col = l16 within n-tile.
    const int b  = row_base >> 8;       // T = 256 rows per batch
    const int t0 = row_base & 255;
    #pragma unroll
    for (int j = 0; j < 12; ++j) {
        const int mat = j >> 2, nt = j & 3;
        const int h = nt * 16 + l16;
        if (mat < 2) {
            _Float16* dst = (mat == 0) ? Qh : Kh;
            #pragma unroll
            for (int r = 0; r < 8; ++r) {
                int row = row_base + r + 8 * hf;
                dst[(size_t)row * HSZ + h] = (_Float16)acc[j][r];
            }
        } else {
            // Vt[b][h][s]; lane's 8 values are consecutive s -> one 16B store
            half8 pv;
            #pragma unroll
            for (int r = 0; r < 8; ++r) pv[r] = (_Float16)acc[j][r];
            *(half8*)(Vt + ((size_t)(b * 64 + h)) * 256 + t0 + 8 * hf) = pv;
        }
    }
}

// ---------------------------------------------------------------------------
// Kernel 3: flash attention. One wave per 16-query tile, key tiles of 32.
// Online softmax in exp2 domain. P transposed C-layout -> A-layout via LDS.
// ---------------------------------------------------------------------------
__device__ __forceinline__ float red_max16(float v) {
    #pragma unroll
    for (int m = 1; m <= 8; m <<= 1) v = fmaxf(v, __shfl_xor(v, m, 32));
    return v;
}
__device__ __forceinline__ float red_sum16(float v) {
    #pragma unroll
    for (int m = 1; m <= 8; m <<= 1) v += __shfl_xor(v, m, 32);
    return v;
}

__global__ __launch_bounds__(128)
void flash_attn_kernel(const _Float16* __restrict__ Qh, const _Float16* __restrict__ Kh,
                       const _Float16* __restrict__ Vt, float* __restrict__ out) {
    __shared__ _Float16 plds[4][16][32];   // per-wave 1KB P transpose slab

    const int lane = threadIdx.x & 31;
    const int wv   = threadIdx.x >> 5;
    const int l16  = lane & 15;
    const int hf   = lane >> 4;
    const int b    = blockIdx.x >> 2;
    const int qt   = (blockIdx.x & 3) + wv * 4;   // interleave for load balance

    const float SC = 0.125f * 1.44269504088896f;  // 1/sqrt(64) * log2(e)

    // Q tile (A layout), resident for the whole loop
    const _Float16* qrow = Qh + ((size_t)(b * TSEQ + qt * 16 + l16)) * HSZ;
    v16h Aq[2];
    #pragma unroll
    for (int kc = 0; kc < 2; ++kc) {
        half8 lo = *(const half8*)(qrow + kc * 32 + hf * 8);
        half8 hi = *(const half8*)(qrow + kc * 32 + hf * 8 + 16);
        Aq[kc] = __builtin_shufflevector(lo, hi,
                     0,1,2,3,4,5,6,7,8,9,10,11,12,13,14,15);
    }

    v8f accO[4] = {};
    float m_i[8], l_i[8];
    #pragma unroll
    for (int r = 0; r < 8; ++r) { m_i[r] = -3.0e38f; l_i[r] = 0.0f; }

    const int ktmax = (qt * 16 + 15) >> 5;
    for (int kt = 0; kt <= ktmax; ++kt) {
        // ---- S = Q K^T : two 16x16 tiles, K-dim = H = 64 -> 2 chunks each
        v8f s[2];
        #pragma unroll
        for (int st = 0; st < 2; ++st) {
            v8f c = {};
            #pragma unroll
            for (int kc = 0; kc < 2; ++kc) {
                // B elem i -> K-index h = kc*32 + (i<8 ? hf*8+i : 16+hf*8+i-8) ; n = key s = tile + l16
                const _Float16* krow = Kh
                    + ((size_t)(b * TSEQ + kt * 32 + st * 16 + l16)) * HSZ
                    + kc * 32 + hf * 8;
                half8 lo = *(const half8*)krow;
                half8 hi = *(const half8*)(krow + 16);
                v16h Bk = __builtin_shufflevector(lo, hi,
                              0,1,2,3,4,5,6,7,8,9,10,11,12,13,14,15);
                c = WMMA_F16(Aq[kc], Bk, c);
            }
            s[st] = c;
        }

        // ---- mask + online softmax (exp2 domain)
        float p0[8], p1[8];
        #pragma unroll
        for (int r = 0; r < 8; ++r) {
            const int row = qt * 16 + r + 8 * hf;
            const int c0  = kt * 32 + l16;
            const int c1  = c0 + 16;
            float v0 = s[0][r] * SC, v1 = s[1][r] * SC;
            v0 = (c0 <= row) ? v0 : -3.0e38f;
            v1 = (c1 <= row) ? v1 : -3.0e38f;
            float tmax = red_max16(fmaxf(v0, v1));
            float mn   = fmaxf(m_i[r], tmax);
            float corr = exp2f(m_i[r] - mn);
            float e0 = exp2f(v0 - mn);
            float e1 = exp2f(v1 - mn);
            l_i[r] = l_i[r] * corr + red_sum16(e0 + e1);
            m_i[r] = mn;
            #pragma unroll
            for (int t4 = 0; t4 < 4; ++t4) accO[t4][r] *= corr;
            p0[r] = e0; p1[r] = e1;
        }

        // ---- transpose P (C layout -> A layout) through per-wave LDS slab
        asm volatile("" ::: "memory");
        #pragma unroll
        for (int r = 0; r < 8; ++r) {
            plds[wv][r + 8 * hf][l16]      = (_Float16)p0[r];
            plds[wv][r + 8 * hf][16 + l16] = (_Float16)p1[r];
        }
        __builtin_amdgcn_wave_barrier();
        asm volatile("" ::: "memory");

        const _Float16* pr = &plds[wv][l16][0];
        half8 plo = *(const half8*)(pr + hf * 8);
        half8 phi = *(const half8*)(pr + 16 + hf * 8);
        v16h Ap = __builtin_shufflevector(plo, phi,
                      0,1,2,3,4,5,6,7,8,9,10,11,12,13,14,15);
        __builtin_amdgcn_wave_barrier();
        asm volatile("" ::: "memory");

        // ---- O += P @ V : Vt[b][h][s] contiguous in s (K-dim of this matmul)
        #pragma unroll
        for (int t4 = 0; t4 < 4; ++t4) {
            const _Float16* vrow = Vt
                + ((size_t)(b * 64 + t4 * 16 + l16)) * 256 + kt * 32 + hf * 8;
            half8 lo = *(const half8*)vrow;
            half8 hi = *(const half8*)(vrow + 16);
            v16h Bv = __builtin_shufflevector(lo, hi,
                          0,1,2,3,4,5,6,7,8,9,10,11,12,13,14,15);
            accO[t4] = WMMA_F16(Ap, Bv, accO[t4]);
        }
    }

    // ---- epilogue: normalize and store f32 output [B][T][64]
    #pragma unroll
    for (int t4 = 0; t4 < 4; ++t4) {
        #pragma unroll
        for (int r = 0; r < 8; ++r) {
            const int row = qt * 16 + r + 8 * hf;
            out[((size_t)(b * TSEQ + row)) * HSZ + t4 * 16 + l16] =
                accO[t4][r] / l_i[r];
        }
    }
}

// ---------------------------------------------------------------------------
// Launch
// ---------------------------------------------------------------------------
extern "C" void kernel_launch(void* const* d_in, const int* in_sizes, int n_in,
                              void* d_out, int out_size, void* d_ws, size_t ws_size,
                              hipStream_t stream) {
    const float* x  = (const float*)d_in[0];
    const float* Wk = (const float*)d_in[1];
    const float* Wq = (const float*)d_in[2];
    const float* Wv = (const float*)d_in[3];
    float* out = (float*)d_out;

    _Float16* ws = (_Float16*)d_ws;
    _Float16* WT = ws + WT_OFF;
    _Float16* Qh = ws + Q_OFF;
    _Float16* Kh = ws + K_OFF;
    _Float16* Vt = ws + V_OFF;

    // 1) weight convert+transpose: 73728 elems
    wt_convert_kernel<<<(int)(WT_ELEM / 256), 256, 0, stream>>>(Wk, Wq, Wv, WT);
    // 2) QKV projection: (B*T)/(16 rows * 8 waves) = 1024 blocks
    qkv_proj_kernel<<<(BATCH * TSEQ) / 128, 256, 0, stream>>>(x, WT, Qh, Kh, Vt);
    // 3) flash attention: B * 4 blocks of 4 waves (16 q-tiles per batch)
    flash_attn_kernel<<<BATCH * 4, 128, 0, stream>>>(Qh, Kh, Vt, out);
}
